// TransformerNet_84464826843160
// MI455X (gfx1250) — compile-verified, matrix operand order checked by reading, not run on hardware
//
#include <hip/hip_runtime.h>
#include <stdint.h>

#define NN    100000
#define NE    1600000
#define INDIM 128
#define HID   64
#define ODIM  16
#define NG    500

typedef __attribute__((ext_vector_type(16))) _Float16 v16h;
typedef __attribute__((ext_vector_type(8)))  float    v8f;
typedef __attribute__((ext_vector_type(8)))  unsigned int v8u;

// ---------- helpers: monotonic int encoding for float atomicMax ----------
__device__ __forceinline__ int ordered_f32(float f) {
    int i = __float_as_int(f);
    return (i >= 0) ? i : (i ^ 0x7FFFFFFF);
}
__device__ __forceinline__ float unordered_f32(int i) {
    return __int_as_float((i >= 0) ? i : (i ^ 0x7FFFFFFF));
}

// ---------- fp32 -> fp16 convert ----------
__global__ void cvt_f32_f16(const float* __restrict__ in, _Float16* __restrict__ out, int count) {
    int t = blockIdx.x * blockDim.x + threadIdx.x;
    if (t < count) out[t] = (_Float16)in[t];
}

// ---------- W[K][64] fp32 -> WT[64][K] fp16 (so B packs as contiguous K-pairs) ----------
__global__ void w_to_f16T(const float* __restrict__ W, _Float16* __restrict__ WT, int K) {
    int t = blockIdx.x * blockDim.x + threadIdx.x;
    if (t >= 64 * K) return;
    int n = t / K, kk = t - n * K;
    WT[t] = (_Float16)W[kk * 64 + n];
}

// ---------- fused 4-projection WMMA GEMM ----------
// Computes {q,k,v,s}[N][64] = A[N][K] x W{q,k,v,s} + b{q,k,v,s} with ONE read of A.
// block = 128 threads (4 waves). Wave w owns output cols [16w,16w+16) of a 16-row strip.
// Per k-step: load A pack once, 4 B packs, issue 4 independent v_wmma_f32_16x16x32_f16.
__global__ void gemm4_n64_wmma(const _Float16* __restrict__ A,
                               const _Float16* __restrict__ BTq, const _Float16* __restrict__ BTk,
                               const _Float16* __restrict__ BTv, const _Float16* __restrict__ BTs,
                               const float* __restrict__ bq, const float* __restrict__ bk,
                               const float* __restrict__ bv, const float* __restrict__ bs,
                               float* __restrict__ q, float* __restrict__ k,
                               float* __restrict__ v, float* __restrict__ so, int K) {
    const int lane = threadIdx.x & 31;
    const int wave = threadIdx.x >> 5;
    const int row0 = blockIdx.x << 4;
    const int c0   = wave << 4;
    const int lg   = lane >> 4;   // lane group 0/1
    const int lm   = lane & 15;
    const _Float16* arow = A + (size_t)(row0 + lm) * K;
    const size_t boff = (size_t)(c0 + lm) * K;
    v8f aq = {}, ak = {}, av = {}, as_ = {};
    for (int kt = 0; kt < K; kt += 32) {
        v8u au, uq, uk, uv, us;
#pragma unroll
        for (int p = 0; p < 8; ++p) {
            // 16-bit operand layout: VGPR p holds K pair at base(0|16) + 8*lanegroup + 2*(p&3)
            const int k0 = kt + ((p & 4) << 2) + (lg << 3) + ((p & 3) << 1);
            au[p] = *(const unsigned int*)(arow + k0);
            uq[p] = *(const unsigned int*)(BTq + boff + k0);
            uk[p] = *(const unsigned int*)(BTk + boff + k0);
            uv[p] = *(const unsigned int*)(BTv + boff + k0);
            us[p] = *(const unsigned int*)(BTs + boff + k0);
        }
        v16h a = __builtin_bit_cast(v16h, au);
        aq  = __builtin_amdgcn_wmma_f32_16x16x32_f16(false, a, false, __builtin_bit_cast(v16h, uq), (short)0, aq,  false, false);
        ak  = __builtin_amdgcn_wmma_f32_16x16x32_f16(false, a, false, __builtin_bit_cast(v16h, uk), (short)0, ak,  false, false);
        av  = __builtin_amdgcn_wmma_f32_16x16x32_f16(false, a, false, __builtin_bit_cast(v16h, uv), (short)0, av,  false, false);
        as_ = __builtin_amdgcn_wmma_f32_16x16x32_f16(false, a, false, __builtin_bit_cast(v16h, us), (short)0, as_, false, false);
    }
    const int col = c0 + lm;
    const float cq = bq[col], ck = bk[col], cv = bv[col], cs = bs[col];
#pragma unroll
    for (int r = 0; r < 8; ++r) {
        // C/D layout: VGPR r -> M = r + 8*lanegroup, N = lane%16
        const size_t o = (size_t)(row0 + r + (lg << 3)) * HID + col;
        q[o]  = aq[r]  + cq;
        k[o]  = ak[r]  + ck;
        v[o]  = av[r]  + cv;
        so[o] = as_[r] + cs;
    }
}

// ---------- per-node init: m = identity(INT_MIN), denom = 0 ----------
__global__ void init_node(int* __restrict__ m_int, float* __restrict__ denom, int n) {
    int t = blockIdx.x * blockDim.x + threadIdx.x;
    if (t < n) { m_int[t] = INT_MIN; denom[t] = 0.0f; }
}

// ---------- per-edge attention scores + segment max (16 lanes per edge, coalesced) ----------
__global__ void edge_dot(const float* __restrict__ q, const float* __restrict__ k,
                         const int* __restrict__ src, const int* __restrict__ dst,
                         float* __restrict__ scores, int* __restrict__ m_int) {
    int t = blockIdx.x * blockDim.x + threadIdx.x;
    int e = t >> 4;
    if (e >= NE) return;
    int sub = t & 15;
    int s = src[e], d = dst[e];
    float4 a = *(const float4*)(q + (size_t)d * HID + (sub << 2));
    float4 b = *(const float4*)(k + (size_t)s * HID + (sub << 2));
    float part = a.x * b.x + a.y * b.y + a.z * b.z + a.w * b.w;
    // butterfly reduction across the 16-lane group (wave32: groups never straddle a wave)
    part += __shfl_xor(part, 1, 32);
    part += __shfl_xor(part, 2, 32);
    part += __shfl_xor(part, 4, 32);
    part += __shfl_xor(part, 8, 32);
    if (sub == 0) {
        float sc = part * 0.125f;   // 1/sqrt(64)
        scores[e] = sc;
        atomicMax(&m_int[d], ordered_f32(sc));
    }
}

// ---------- exp(score - max) + segment sum ----------
__global__ void edge_exp(float* __restrict__ scores, const int* __restrict__ dst,
                         const int* __restrict__ m_int, float* __restrict__ denom) {
    int e = blockIdx.x * blockDim.x + threadIdx.x;
    if (e >= NE) return;
    int d = dst[e];
    int mi = m_int[d];
    float m = (mi == INT_MIN) ? 0.0f : unordered_f32(mi);
    float ex = __expf(scores[e] - m);
    scores[e] = ex;
    atomicAdd(&denom[d], ex);
}

// ---------- scatter alpha * v[src] into hnext[dst]; 16 lanes per edge (coalesced) ----------
__global__ void edge_scatter(const float* __restrict__ evals, const float* __restrict__ denom,
                             const float* __restrict__ v, const int* __restrict__ src,
                             const int* __restrict__ dst, float* __restrict__ hnext) {
    int t = blockIdx.x * blockDim.x + threadIdx.x;
    if (t >= NE * (HID / 4)) return;
    int e = t >> 4;
    int c = (t & 15) << 2;
    int s = src[e], d = dst[e];
    float alpha = evals[e] / (denom[d] + 1e-16f);
    float4 vv = *(const float4*)(v + (size_t)s * HID + c);
    float* o = hnext + (size_t)d * HID + c;
    atomicAdd(o + 0, alpha * vv.x);
    atomicAdd(o + 1, alpha * vv.y);
    atomicAdd(o + 2, alpha * vv.z);
    atomicAdd(o + 3, alpha * vv.w);
}

// ---------- relu in place + fp16 copy for next layer's WMMA ----------
__global__ void relu_store(float* __restrict__ h, _Float16* __restrict__ hf16, int count) {
    int t = blockIdx.x * blockDim.x + threadIdx.x;
    if (t >= count) return;
    float x = fmaxf(h[t], 0.0f);
    h[t] = x;
    hf16[t] = (_Float16)x;
}

__global__ void zero_f32(float* __restrict__ p, int count) {
    int t = blockIdx.x * blockDim.x + threadIdx.x;
    if (t < count) p[t] = 0.0f;
}

// ---------- global_add_pool ----------
__global__ void pool_kernel(const float* __restrict__ h, const int* __restrict__ batch,
                            float* __restrict__ pooled) {
    int t = blockIdx.x * blockDim.x + threadIdx.x;
    if (t >= NN * (HID / 4)) return;
    int n = t >> 4;
    int c = (t & 15) << 2;
    int g = batch[n];
    float4 vv = *(const float4*)(h + (size_t)n * HID + c);
    float* o = pooled + (size_t)g * HID + c;
    atomicAdd(o + 0, vv.x);
    atomicAdd(o + 1, vv.y);
    atomicAdd(o + 2, vv.z);
    atomicAdd(o + 3, vv.w);
}

// ---------- MLP head: relu(pooled@W1+b1)@W2+b2, one block per graph ----------
__global__ void mlp_head(const float* __restrict__ pooled,
                         const float* __restrict__ W1, const float* __restrict__ b1,
                         const float* __restrict__ W2, const float* __restrict__ b2,
                         float* __restrict__ out) {
    __shared__ float hid[HID];
    int g = blockIdx.x, t = threadIdx.x;
    float acc = b1[t];
    for (int kk = 0; kk < HID; ++kk) acc = fmaf(pooled[g * HID + kk], W1[kk * HID + t], acc);
    hid[t] = fmaxf(acc, 0.0f);
    __syncthreads();
    if (t < ODIM) {
        float o = b2[t];
        for (int kk = 0; kk < HID; ++kk) o = fmaf(hid[kk], W2[kk * ODIM + t], o);
        out[g * ODIM + t] = o;
    }
}

// ---------------- host-side layer driver ----------------
static void run_layer(const _Float16* hf16_in, int K,
                      const float* Wq, const float* bq, const float* Wk, const float* bk,
                      const float* Wv, const float* bv, const float* Ws, const float* bs,
                      const int* src, const int* dst,
                      float* q, float* k, float* v, float* hnext,
                      _Float16* wtq, _Float16* wtk, _Float16* wtv, _Float16* wts,
                      float* scores, int* m_int, float* denom,
                      _Float16* hf16_out, hipStream_t stream) {
    const int ROWB = NN / 16;                 // 6250 row-tiles
    const int wtB  = (64 * K + 255) / 256;
    // stage transposed f16 weights for all four projections, then one fused GEMM
    w_to_f16T<<<wtB, 256, 0, stream>>>(Wq, wtq, K);
    w_to_f16T<<<wtB, 256, 0, stream>>>(Wk, wtk, K);
    w_to_f16T<<<wtB, 256, 0, stream>>>(Wv, wtv, K);
    w_to_f16T<<<wtB, 256, 0, stream>>>(Ws, wts, K);
    gemm4_n64_wmma<<<ROWB, 128, 0, stream>>>(hf16_in, wtq, wtk, wtv, wts,
                                             bq, bk, bv, bs, q, k, v, hnext, K);
    // attention: per-dst softmax over edges + weighted aggregation
    init_node<<<(NN + 255) / 256, 256, 0, stream>>>(m_int, denom, NN);
    edge_dot<<<(NE * 16) / 256, 256, 0, stream>>>(q, k, src, dst, scores, m_int);
    edge_exp<<<NE / 256, 256, 0, stream>>>(scores, dst, m_int, denom);
    edge_scatter<<<(NE * 16) / 256, 256, 0, stream>>>(scores, denom, v, src, dst, hnext);
    relu_store<<<(NN * HID) / 256, 256, 0, stream>>>(hnext, hf16_out, NN * HID);
}

extern "C" void kernel_launch(void* const* d_in, const int* in_sizes, int n_in,
                              void* d_out, int out_size, void* d_ws, size_t ws_size,
                              hipStream_t stream) {
    const float* x   = (const float*)d_in[0];
    const float* Wq0 = (const float*)d_in[1];  const float* bq0 = (const float*)d_in[2];
    const float* Wk0 = (const float*)d_in[3];  const float* bk0 = (const float*)d_in[4];
    const float* Wv0 = (const float*)d_in[5];  const float* bv0 = (const float*)d_in[6];
    const float* Ws0 = (const float*)d_in[7];  const float* bs0 = (const float*)d_in[8];
    const float* Wqs = (const float*)d_in[9];  const float* bqs = (const float*)d_in[10];
    const float* Wks = (const float*)d_in[11]; const float* bks = (const float*)d_in[12];
    const float* Wvs = (const float*)d_in[13]; const float* bvs = (const float*)d_in[14];
    const float* Wss = (const float*)d_in[15]; const float* bss = (const float*)d_in[16];
    const float* W1  = (const float*)d_in[17]; const float* b1  = (const float*)d_in[18];
    const float* W2  = (const float*)d_in[19]; const float* b2  = (const float*)d_in[20];
    const int* edge_index = (const int*)d_in[21];
    const int* batch      = (const int*)d_in[22];
    const int* src = edge_index;
    const int* dst = edge_index + NE;
    float* out = (float*)d_out;

    // ---- carve workspace (256B aligned) ----
    uint8_t* p = (uint8_t*)d_ws;
    auto carve = [&](size_t bytes) {
        uint8_t* r = p;
        p += (bytes + 255) & ~(size_t)255;
        return r;
    };
    float*    q      = (float*)carve((size_t)NN * HID * 4);
    float*    k      = (float*)carve((size_t)NN * HID * 4);
    float*    v      = (float*)carve((size_t)NN * HID * 4);
    float*    hnext  = (float*)carve((size_t)NN * HID * 4);
    float*    scores = (float*)carve((size_t)NE * 4);
    float*    denom  = (float*)carve((size_t)NN * 4);
    int*      m_int  = (int*)carve((size_t)NN * 4);
    float*    pooled = (float*)carve((size_t)NG * HID * 4);
    _Float16* hf16   = (_Float16*)carve((size_t)NN * INDIM * 2);
    _Float16* wtq    = (_Float16*)carve((size_t)64 * INDIM * 2);
    _Float16* wtk    = (_Float16*)carve((size_t)64 * INDIM * 2);
    _Float16* wtv    = (_Float16*)carve((size_t)64 * INDIM * 2);
    _Float16* wts    = (_Float16*)carve((size_t)64 * INDIM * 2);

    // layer 0: IN_DIM -> HID
    cvt_f32_f16<<<(NN * INDIM + 255) / 256, 256, 0, stream>>>(x, hf16, NN * INDIM);
    run_layer(hf16, INDIM, Wq0, bq0, Wk0, bk0, Wv0, bv0, Ws0, bs0,
              src, dst, q, k, v, hnext, wtq, wtk, wtv, wts,
              scores, m_int, denom, hf16, stream);
    // layers 1..2: HID -> HID (stacked weights)
    for (int li = 0; li < 2; ++li) {
        run_layer(hf16, HID,
                  Wqs + li * HID * HID, bqs + li * HID,
                  Wks + li * HID * HID, bks + li * HID,
                  Wvs + li * HID * HID, bvs + li * HID,
                  Wss + li * HID * HID, bss + li * HID,
                  src, dst, q, k, v, hnext, wtq, wtk, wtv, wts,
                  scores, m_int, denom, hf16, stream);
    }
    // pooling + MLP head
    zero_f32<<<(NG * HID + 255) / 256, 256, 0, stream>>>(pooled, NG * HID);
    pool_kernel<<<(NN * 16) / 256, 256, 0, stream>>>(hnext, batch, pooled);
    mlp_head<<<NG, HID, 0, stream>>>(pooled, W1, b1, W2, b2, out);
}
